// FullyConnectedSteerableGeometricProductLayer_89043261980956
// MI455X (gfx1250) — compile-verified, matrix operand order checked by reading, not run on hardware
//
#include <hip/hip_runtime.h>
#include <hip/hip_bf16.h>
#include <hip/hip_fp16.h>
#include <stdint.h>

// MI455X (gfx1250, wave32). Compile-only target.
//
// Roofline: 512 MB fp16 output stores dominate (~22us @ 23.3 TB/s); phase-1
// weight streaming adds 128 MB (~5.5us). Compute (~5.4 GFLOP) is negligible,
// so the design centers on coalesced b128 traffic: async global->LDS staging
// of the k tile (ASYNCcnt), v_wmma_f32_16x16x32_f16 for the pair product, and
// non-temporal b128 output stores.

typedef __attribute__((ext_vector_type(16))) _Float16 v16h;
typedef __attribute__((ext_vector_type(8)))  _Float16 v8h;
typedef __attribute__((ext_vector_type(8)))  float    v8f;
typedef __attribute__((ext_vector_type(4)))  unsigned v4u;

#define GS 2048   // sequence length S
#define GB 8      // batch B

// ---------------------------------------------------------------------------
// Kernel 1: MVLinear for q (blockIdx.y==0) and k (blockIdx.y==1).
//   out[b,n,i] = sum_m x[b,m,i] * w[n,m,grade(i)]  (+ bias[n] on i==0)
// One block = 32 n values; x chunk staged in LDS as xs[m][i][b] so the 8-way
// batch accumulation per thread is two ds_load_b128. Each 16B w line
// (4 grades) is consumed exactly once.
// ---------------------------------------------------------------------------
__global__ void __launch_bounds__(256)
mvlinear_kernel(const float* __restrict__ x,
                const float* __restrict__ wq, const float* __restrict__ bq,
                const float* __restrict__ wk, const float* __restrict__ bk,
                _Float16* __restrict__ qh, _Float16* __restrict__ kh) {
  const float* w; const float* bias; _Float16* out;
  if (blockIdx.y == 0) { w = wq; bias = bq; out = qh; }
  else                 { w = wk; bias = bk; out = kh; }

  __shared__ float xs[128][8][8];   // [m_local][blade i][batch b] = 32 KB

  const int tid = threadIdx.x;
  const int n_local = tid >> 3;     // 0..31
  const int i = tid & 7;            // blade component
  // blade order [1,e1,e2,e3,e12,e13,e23,e123] -> grades [0,1,1,1,2,2,2,3]
  const int g = (i == 0) ? 0 : (i < 4) ? 1 : (i < 7) ? 2 : 3;
  const int n = blockIdx.x * 32 + n_local;

  float acc[8];
  #pragma unroll
  for (int b = 0; b < 8; ++b) acc[b] = 0.f;

  for (int c = 0; c < GS / 128; ++c) {
    const int m0 = c * 128;
    // cooperative stage: x[b][m0+ml][i] -> xs[ml][i][b]
    #pragma unroll
    for (int r = 0; r < 8; ++r) {
      int idx = tid + 256 * r;                   // float4 index 0..2047
      int ml = idx >> 4, b = (idx >> 1) & 7, i4 = (idx & 1) * 4;
      const float4 v = *(const float4*)(x + (((size_t)b * GS) + m0 + ml) * 8 + i4);
      xs[ml][i4 + 0][b] = v.x; xs[ml][i4 + 1][b] = v.y;
      xs[ml][i4 + 2][b] = v.z; xs[ml][i4 + 3][b] = v.w;
    }
    __syncthreads();
    const float* wp = w + ((size_t)n * GS + m0) * 4 + g;
    for (int ml = 0; ml < 128; ++ml) {
      const float wv = wp[(size_t)ml * 4];
      const float4 xa = *(const float4*)&xs[ml][i][0];
      const float4 xb = *(const float4*)&xs[ml][i][4];
      acc[0] += wv * xa.x; acc[1] += wv * xa.y;
      acc[2] += wv * xa.z; acc[3] += wv * xa.w;
      acc[4] += wv * xb.x; acc[5] += wv * xb.y;
      acc[6] += wv * xb.z; acc[7] += wv * xb.w;
    }
    __syncthreads();
  }
  if (i == 0) {
    const float bv = bias[n];
    #pragma unroll
    for (int b = 0; b < 8; ++b) acc[b] += bv;
  }
  #pragma unroll
  for (int b = 0; b < 8; ++b)
    out[((size_t)b * GS + n) * 8 + i] = (_Float16)acc[b];
}

// ---------------------------------------------------------------------------
// Kernel 2: qc[b,s,j,k] = sum_i q[b,s,i] * C[i,j,k]  (fp16 out, 2 MB)
// One thread per (b,s); Cayley (2 KB) cached in LDS; 128 B contiguous stores.
// ---------------------------------------------------------------------------
__global__ void __launch_bounds__(256)
qc_kernel(const _Float16* __restrict__ qh, const float* __restrict__ cayley,
          _Float16* __restrict__ qc) {
  __shared__ float cs[512];
  const int t = threadIdx.x;
  cs[t]       = cayley[t];
  cs[t + 256] = cayley[t + 256];
  __syncthreads();

  const int gid = blockIdx.x * 256 + t;          // flat (b*S + s)
  const v8h q8 = *(const v8h*)(qh + (size_t)gid * 8);
  float qv[8];
  #pragma unroll
  for (int e = 0; e < 8; ++e) qv[e] = (float)q8[e];

  v8h* dst = (v8h*)(qc + (size_t)gid * 64);
  #pragma unroll
  for (int j = 0; j < 8; ++j) {
    v8h row;
    #pragma unroll
    for (int k = 0; k < 8; ++k) {
      float a = 0.f;
      #pragma unroll
      for (int i = 0; i < 8; ++i) a += qv[i] * cs[i * 64 + j * 8 + k];
      row[k] = (_Float16)a;
    }
    dst[j] = row;
  }
}

// ---------------------------------------------------------------------------
// Kernel 3: out[b,s,t,j] = sum_k qc[b,s,j,k] * k[b,t,k]   (the 512 MB stream)
// Block = 8 waves, all with the same batch b. k[b] (32 KB) is staged once per
// block into LDS via global_load_async_to_lds_b128 (ASYNCcnt), then:
// per wave: one (b, s-pair); A (16x32 f16, M = s_local*8+j, K zero-padded
// past 8) loaded once; loop 128 t-tiles, each = 1 ds_load_b128 + branchless
// cndmask zeroing of the K>=8 half + 1 WMMA + 1 NT b128 store per lane.
// f32 C/D layout puts all 8 j-components of one (s,t) in a single lane ->
// contiguous 16 B store, 256 B across lanes 0-15.
// ---------------------------------------------------------------------------
__global__ void __launch_bounds__(256)
geoprod_kernel(const _Float16* __restrict__ qc, const _Float16* __restrict__ kh,
               _Float16* __restrict__ out) {
  __shared__ __align__(16) _Float16 ksh[GS * 8];   // 32 KB: k[b] for this block

  const int lane = threadIdx.x & 31;
  const int wave = (int)((blockIdx.x * blockDim.x + threadIdx.x) >> 5);
  const int sp = wave & ((GS / 2) - 1);            // s-pair 0..1023
  const int b  = wave >> 10;                       // 0..7 (block-uniform)
  const int s_base = sp * 2;

  // ---- stage k[b] (32 KB) into LDS with async global->LDS DMA (ASYNCcnt) ---
  // The VDST operand is derived from the ksh pointer itself (ptrtoint): this
  // escapes the LDS object so the "memory" clobber is known to cover it (the
  // round-2 collapse), and the low 32 bits of the generic address ARE the DS
  // byte offset per the ISA aperture mapping.
  {
    const _Float16* kb = kh + (size_t)b * GS * 8;
    #pragma unroll
    for (int r = 0; r < 8; ++r) {
      const int idx = (int)threadIdx.x + 256 * r;      // row 0..2047
      const unsigned ldsoff = (unsigned)(uintptr_t)(ksh + (size_t)idx * 8);
      const _Float16* gsrc = kb + (size_t)idx * 8;
      asm volatile("global_load_async_to_lds_b128 %0, %1, off"
                   :: "v"(ldsoff), "v"(gsrc) : "memory");
    }
    asm volatile("s_wait_asynccnt 0" ::: "memory");
    __syncthreads();
  }

  // A: lanes 0-15 hold K=0..7 (VGPRs 0-3); lanes 16-31 / VGPRs 4-7 = K pad.
  v16h A = {};
  if (lane < 16) {
    const int s = s_base + (lane >> 3);
    const int j = lane & 7;
    const v8h q8 = *(const v8h*)(qc + (((size_t)b * GS + s) * 8 + j) * 8);
    #pragma unroll
    for (int e = 0; e < 8; ++e) A[e] = q8[e];
  }

  union BU { v16h h; v4u d[2]; };
  const v4u zero4 = {0u, 0u, 0u, 0u};
  const bool lo16 = lane < 16;
  const int s_off = lane >> 4;                     // 0 or 1
  const int tcol  = lane & 15;
  _Float16* ob = out + ((size_t)b * GS + s_base) * (size_t)GS * 8;

  for (int tt = 0; tt < GS / 16; ++tt) {
    const int t_base = tt * 16;
    // B: 32x16 f16; lanes 0-15 (N=lane) need K=0..7 in VGPRs 0-3; rest zero.
    const v4u kw = *(const v4u*)(ksh + (size_t)(t_base + tcol) * 8);
    BU Bm;
    #pragma unroll
    for (int e = 0; e < 4; ++e) Bm.d[0][e] = lo16 ? kw[e] : 0u;
    Bm.d[1] = zero4;

    v8f C = {};
    C = __builtin_amdgcn_wmma_f32_16x16x32_f16(
            /*neg_a=*/false, A, /*neg_b=*/false, Bm.h,
            /*c_mod=*/(short)0, C, /*reuse_a=*/false, /*reuse_b=*/false);

    // C layout: VGPR v -> M = v + 8*(lane>=16) = s_local*8 + j, N = lane%16.
    v8h o;
    #pragma unroll
    for (int e = 0; e < 8; ++e) o[e] = (_Float16)C[e];
    __builtin_nontemporal_store(
        o, (v8h*)(ob + ((size_t)s_off * GS + (t_base + tcol)) * 8));
  }
}

// ---------------------------------------------------------------------------
extern "C" void kernel_launch(void* const* d_in, const int* in_sizes, int n_in,
                              void* d_out, int out_size, void* d_ws, size_t ws_size,
                              hipStream_t stream) {
  (void)in_sizes; (void)n_in; (void)out_size; (void)ws_size;
  const float* x      = (const float*)d_in[0];
  const float* wq     = (const float*)d_in[1];
  const float* bq     = (const float*)d_in[2];
  const float* wk     = (const float*)d_in[3];
  const float* bk     = (const float*)d_in[4];
  const float* cayley = (const float*)d_in[5];

  // Workspace layout (2.5 MB total):
  //   qh: [B,S,8] fp16 (256 KB), kh: same (256 KB), qc: [B,S,8,8] fp16 (2 MB)
  _Float16* qh = (_Float16*)d_ws;
  _Float16* kh = (_Float16*)((char*)d_ws + (size_t)GB * GS * 8 * 2);
  _Float16* qc = (_Float16*)((char*)d_ws + (size_t)GB * GS * 8 * 2 * 2);
  // Reference output dtype is float16 -> treat d_out as fp16.
  _Float16* out = (_Float16*)d_out;

  dim3 g1(GS / 32, 2);
  mvlinear_kernel<<<g1, 256, 0, stream>>>(x, wq, bq, wk, bk, qh, kh);
  qc_kernel<<<(GB * GS) / 256, 256, 0, stream>>>(qh, cayley, qc);
  geoprod_kernel<<<(GB * (GS / 2)) / 8, 256, 0, stream>>>(qc, kh, out);
}
